// FLASH_ShareA_FFConvM_76759655514382
// MI455X (gfx1250) — matrix operand order repaired and checked by reading, not compiled, not run on hardware
//
#include <hip/hip_runtime.h>
#include <hip/hip_bf16.h>
#include <stdint.h>

// Problem constants (B, SEQ, DIM) = (4, 8192, 512), G=32, n=256, QK=128, HID=512
#define B_    4
#define SEQ_  8192
#define DIM_  512
#define G_    32
#define NG_   256          // SEQ/G
#define QK_   128
#define HID_  512
#define EPS_  1e-5f

#define USE_ASYNC_LDS 1

typedef __bf16 bf16;
typedef __bf16 v16bf __attribute__((ext_vector_type(16)));
typedef __bf16 v8bf  __attribute__((ext_vector_type(8)));
typedef float  v8f   __attribute__((ext_vector_type(8)));

__device__ __forceinline__ v8f wmma_bf16(v16bf a, v16bf b, v8f c) {
  return __builtin_amdgcn_wmma_f32_16x16x32_bf16(false, a, false, b, (short)0, c, false, false);
}

__device__ __forceinline__ float silu_f(float v) { return v / (1.f + __expf(-v)); }

// combine two contiguous 8-element runs into one fragment register set
__device__ __forceinline__ v16bf frag_cat(const bf16* lo, const bf16* hi) {
  v8bf a = *reinterpret_cast<const v8bf*>(lo);
  v8bf b = *reinterpret_cast<const v8bf*>(hi);
  v16bf r;
#pragma unroll
  for (int e = 0; e < 8; ++e) { r[e] = a[e]; r[8 + e] = b[e]; }
  return r;
}

// ---------------------------------------------------- fp32 -> bf16 transpose
// src: K x N row-major f32;  dst: N x K row-major bf16 (i.e. src^T)
__global__ void k_cvt_t(const float* __restrict__ s, bf16* __restrict__ d, int K, int N) {
  int i = blockIdx.x * 256 + threadIdx.x;
  if (i < K * N) {
    int k = i / N, n = i - k * N;
    d[(long)n * K + k] = (bf16)s[i];
  }
}

// ---------------------------------------------------- LayerNorm + seq shift
__global__ __launch_bounds__(256)
void k_ln_shift(const float* __restrict__ x, const float* __restrict__ lg,
                const float* __restrict__ lb, float* __restrict__ xs_f,
                bf16* __restrict__ xs_b) {
  long t = blockIdx.x;
  int pos = (int)(t & (SEQ_ - 1));
  const float* xr = x + t * DIM_;
  int tid = threadIdx.x;
  float v0 = xr[tid], v1 = xr[tid + 256];
  float s = v0 + v1, s2 = v0 * v0 + v1 * v1;
  for (int o = 16; o; o >>= 1) { s += __shfl_down(s, o, 32); s2 += __shfl_down(s2, o, 32); }
  __shared__ float ls[8], ls2[8];
  int wave = tid >> 5, lane = tid & 31;
  if (lane == 0) { ls[wave] = s; ls2[wave] = s2; }
  __syncthreads();
  if (wave == 0) {
    s = (lane < 8) ? ls[lane] : 0.f;
    s2 = (lane < 8) ? ls2[lane] : 0.f;
    for (int o = 4; o; o >>= 1) { s += __shfl_down(s, o, 32); s2 += __shfl_down(s2, o, 32); }
    if (lane == 0) { ls[0] = s; ls2[0] = s2; }
  }
  __syncthreads();
  float mu = ls[0] * (1.f / DIM_);
  float var = ls2[0] * (1.f / DIM_) - mu * mu;
  float rstd = rsqrtf(var + EPS_);
#pragma unroll
  for (int q = 0; q < 2; ++q) {
    int c = tid + q * 256;
    float v = q ? v1 : v0;
    float y = (v - mu) * rstd * lg[c] + lb[c];
    if (c >= 256) {
      long o = t * DIM_ + c;
      xs_f[o] = y; xs_b[o] = (bf16)y;
    } else {
      if (pos + 1 < SEQ_) { long o = (t + 1) * DIM_ + c; xs_f[o] = y; xs_b[o] = (bf16)y; }
      if (pos == 0)       { long o = t * DIM_ + c;       xs_f[o] = 0.f; xs_b[o] = (bf16)0.f; }
    }
  }
}

// ------------------------------------------------------- generic WMMA GEMM
// C[M,N] = A[M,K] * Bt^T  (Bt is N x K row-major = B transposed), 128x128 tile.
// mode 0: outF = v; mode 1: hidden/gate split (+ transposed hidden copy);
// mode 2: outF = silu(v); mode 3: outFinal = v*aux0 + aux1
__global__ __launch_bounds__(256)
void k_gemm_bf16(const bf16* __restrict__ A0, const bf16* __restrict__ A1, int kSplit,
                 int lda0, int lda1, const bf16* __restrict__ Bt,
                 int N, int K, int mode,
                 float* __restrict__ outF, bf16* __restrict__ outH,
                 bf16* __restrict__ outHT, float* __restrict__ outG,
                 const float* __restrict__ aux0, const float* __restrict__ aux1,
                 float* __restrict__ outFinal) {
  __shared__ __align__(16) bf16 As[128 * 32];   // row-major [row][32]
  __shared__ __align__(16) bf16 Bst[128 * 32];  // col-major tile [col][32]
  int tid = threadIdx.x;
  int wave = tid >> 5, lane = tid & 31;
  int waveM = wave >> 1, waveN = wave & 1;
  int h = lane >> 4, ml = lane & 15;
  long tileM = (long)blockIdx.y * 128;
  int tileN = blockIdx.x * 128;
  v8f acc[2][4] = {};
  for (int k0 = 0; k0 < K; k0 += 32) {
    // A tile 128x32: 512 x 16B chunks, 2/thread
#pragma unroll
    for (int i = 0; i < 2; ++i) {
      int c = tid + i * 256;
      int row = c >> 2, seg = c & 3;
      int gk = k0 + seg * 8;
      const bf16* src = (gk < kSplit) ? (A0 + (tileM + row) * (long)lda0 + gk)
                                      : (A1 + (tileM + row) * (long)lda1 + (gk - kSplit));
      *reinterpret_cast<uint4*>(&As[row * 32 + seg * 8]) =
          *reinterpret_cast<const uint4*>(src);
    }
    // B tile (transposed source): 128 cols x 32 k, contiguous per col
#pragma unroll
    for (int i = 0; i < 2; ++i) {
      int c = tid + i * 256;
      int col = c >> 2, seg = c & 3;
      const bf16* src = Bt + (long)(tileN + col) * K + k0 + seg * 8;
      *reinterpret_cast<uint4*>(&Bst[col * 32 + seg * 8]) =
          *reinterpret_cast<const uint4*>(src);
    }
    if (k0 + 32 < K)
      __builtin_prefetch(Bt + (long)tileN * K + k0 + 32, 0, 1);
    __syncthreads();
    v16bf af[2];
#pragma unroll
    for (int rt = 0; rt < 2; ++rt) {
      const bf16* r = &As[(waveM * 32 + rt * 16 + ml) * 32];
      af[rt] = frag_cat(r + 16 * h, r + 16 * h + 8);   // A 16x32, k-contig per lane half
    }
#pragma unroll
    for (int ct = 0; ct < 4; ++ct) {
      const bf16* cptr = &Bst[(waveN * 64 + ct * 16 + ml) * 32 + 16 * h];
      v16bf bfr = frag_cat(cptr, cptr + 8);           // B 32x16, k-contig per lane half
#pragma unroll
      for (int rt = 0; rt < 2; ++rt) acc[rt][ct] = wmma_bf16(af[rt], bfr, acc[rt][ct]);
    }
    __syncthreads();
  }
#pragma unroll
  for (int rt = 0; rt < 2; ++rt)
#pragma unroll
    for (int ct = 0; ct < 4; ++ct)
#pragma unroll
      for (int e = 0; e < 8; ++e) {
        long row = tileM + waveM * 32 + rt * 16 + 8 * h + e;
        int col = tileN + waveN * 64 + ct * 16 + ml;
        float v = acc[rt][ct][e];
        if (mode == 0) outF[row * N + col] = v;
        else if (mode == 1) {
          if (col < 512) {
            outH[row * 512 + col] = (bf16)v;
            // transposed per-group copy for attention values:
            // hidT[(b*G+g)*512 + col][rowInGroup]
            outHT[((row >> 8) * 512 + col) * 256L + (row & 255)] = (bf16)v;
          } else outG[row * 512 + (col - 512)] = v;
        } else if (mode == 2) {
          outF[row * N + col] = silu_f(v);
        } else {
          long o = row * 512 + col;
          outFinal[o] = v * aux0[o] + aux1[o];
        }
      }
}

// ------------------------------------------------------------- qk group mean
__global__ void k_qkmean(const float* __restrict__ qk, float* __restrict__ qkm) {
  int bg = blockIdx.x;
  int c = threadIdx.x;
  long base = (long)bg * NG_ * QK_;
  float s = 0.f;
  for (int i = 0; i < NG_; ++i) s += qk[base + (long)i * QK_ + c];
  qkm[(long)bg * QK_ + c] = s * (1.f / NG_);
}

// --------------------------------------------------------------- attention
// one block per (b, g, head).  LDS: P 256x256 bf16 (128KB) + union{Q+Kt | Ht} 64KB
__global__ __launch_bounds__(256)
void k_attn(const float* __restrict__ qk, const float* __restrict__ qkm,
            const float* __restrict__ g4, const float* __restrict__ b4,
            const float* __restrict__ g2, const float* __restrict__ b2,
            const bf16* __restrict__ hidT, const float* __restrict__ gateF,
            bf16* __restrict__ attnO) {
  extern __shared__ __align__(16) char smem[];
  bf16* P  = (bf16*)smem;                    // 256*256
  bf16* Q  = (bf16*)(smem + 256 * 256 * 2);  // 256*64
  bf16* Kt = Q + 256 * 64;                   // 256*64
  bf16* Ht = (bf16*)(smem + 256 * 256 * 2);  // 128 cols x 256 rows, reuses Q/Kt
  __shared__ float sqA[64], sqB[64], sqZ[64], skA[64], skB[64], skZ[64];
  int tid = threadIdx.x;
  int id = blockIdx.x;
  int b = id >> 7;
  int gg = (id >> 2) & 31;
  int hh = id & 3;
  int srcS = hh >> 1;
  int half = hh & 1;
  if (tid < 64) {
    int C = half * 64 + tid;
    float m = qkm[((long)b * G_ + gg) * QK_ + C];
    float qoff = m * g4[0 * QK_ + C] + b4[0 * QK_ + C];
    float koff = m * g4[1 * QK_ + C] + b4[1 * QK_ + C];
    float qsc  = m * g4[2 * QK_ + C] + b4[2 * QK_ + C];
    float ksc  = m * g4[3 * QK_ + C] + b4[3 * QK_ + C];
    if (srcS == 0) {
      sqA[tid] = qsc; sqB[tid] = qoff; sqZ[tid] = 0.f;
      skA[tid] = ksc; skB[tid] = koff; skZ[tid] = 0.f;
    } else {
      float qso = m * g2[0 * QK_ + C] + b2[0 * QK_ + C];
      float kso = m * g2[1 * QK_ + C] + b2[1 * QK_ + C];
      sqA[tid] = qsc * qsc; sqB[tid] = qoff * qsc + qso; sqZ[tid] = qso;
      skA[tid] = ksc * ksc; skB[tid] = koff * ksc + kso; skZ[tid] = kso;
    }
  }
  __syncthreads();
  {  // stage Q, K (256x64) as bf16, vectorized stores
    int row = tid;
    int srow = srcS ? row - 1 : row;
    long rbase = ((long)b * SEQ_ + gg * NG_ + srow) * QK_ + half * 64;
#pragma unroll
    for (int cc = 0; cc < 64; cc += 8) {
      v8bf qv8, kv8;
      if (srcS && row == 0) {
#pragma unroll
        for (int e = 0; e < 8; ++e) { qv8[e] = (bf16)sqZ[cc + e]; kv8[e] = (bf16)skZ[cc + e]; }
      } else {
#pragma unroll
        for (int e = 0; e < 8; ++e) {
          float xv = qk[rbase + cc + e];
          qv8[e] = (bf16)(xv * sqA[cc + e] + sqB[cc + e]);
          kv8[e] = (bf16)(xv * skA[cc + e] + skB[cc + e]);
        }
      }
      *reinterpret_cast<v8bf*>(&Q[row * 64 + cc]) = qv8;
      *reinterpret_cast<v8bf*>(&Kt[row * 64 + cc]) = kv8;
    }
  }
  __syncthreads();
  int wave = tid >> 5, lane = tid & 31, h = lane >> 4, ml = lane & 15;
  int r0 = wave * 32;
  const float sc = 0.17677669529663687f;  // HD^-0.5 = 32^-0.5
  // S = Q*K^T -> masked scaled logits in P
  for (int ct = 0; ct < 16; ++ct) {
    for (int rt = 0; rt < 2; ++rt) {
      v8f a = {};
#pragma unroll
      for (int ks = 0; ks < 2; ++ks) {
        const bf16* qp = &Q[(r0 + rt * 16 + ml) * 64 + ks * 32];
        v16bf af = frag_cat(qp + 16 * h, qp + 16 * h + 8);
        const bf16* kp = &Kt[(ct * 16 + ml) * 64 + ks * 32 + 16 * h];
        v16bf bfr = frag_cat(kp, kp + 8);
        a = wmma_bf16(af, bfr, a);
      }
#pragma unroll
      for (int e = 0; e < 8; ++e) {
        int grow = r0 + rt * 16 + 8 * h + e;
        int gcol = ct * 16 + ml;
        float v = (gcol > grow) ? -3.0e38f : a[e] * sc;
        P[grow * 256 + gcol] = (bf16)v;
      }
    }
  }
  __syncthreads();
  {  // row softmax, vectorized (8 bf16 per LDS op)
    int row = tid;
    v8bf* rp = reinterpret_cast<v8bf*>(&P[row * 256]);
    float mx = -3.4e38f;
    for (int j = 0; j < 32; ++j) {
      v8bf v = rp[j];
#pragma unroll
      for (int e = 0; e < 8; ++e) mx = fmaxf(mx, (float)v[e]);
    }
    float s = 0.f;
    for (int j = 0; j < 32; ++j) {
      v8bf v = rp[j];
#pragma unroll
      for (int e = 0; e < 8; ++e) {
        float ev = __expf((float)v[e] - mx);
        s += ev;
        v[e] = (bf16)ev;
      }
      rp[j] = v;
    }
    float inv = 1.f / s;
    for (int j = 0; j < 32; ++j) {
      v8bf v = rp[j];
#pragma unroll
      for (int e = 0; e < 8; ++e) v[e] = (bf16)((float)v[e] * inv);
      rp[j] = v;
    }
  }
  __syncthreads();
  long gRowBase = (long)b * SEQ_ + gg * NG_;
  long tBase = ((long)(b * G_ + gg) * 512) * 256;   // hidT[(bg*512 + col)*256 + row]
  for (int cg = 0; cg < 4; ++cg) {
    // stage value chunk transposed: Ht[col][256], contiguous 16B chunks -> async DMA
#pragma unroll
    for (int i = 0; i < 16; ++i) {
      int c = tid + i * 256;        // 4096 chunks: 128 cols x 32 row-segs
      int col = c >> 5, rseg = c & 31;
      const bf16* gp = hidT + tBase + (long)(cg * 128 + col) * 256 + rseg * 8;
#if USE_ASYNC_LDS
      unsigned la = (unsigned)(uintptr_t)&Ht[col * 256 + rseg * 8];
      asm volatile("global_load_async_to_lds_b128 %0, %1, off" :: "v"(la), "v"(gp) : "memory");
#else
      *reinterpret_cast<uint4*>(&Ht[col * 256 + rseg * 8]) =
          *reinterpret_cast<const uint4*>(gp);
#endif
    }
#if USE_ASYNC_LDS
    asm volatile("s_wait_asynccnt 0x0" ::: "memory");
#endif
    __syncthreads();
    v8f acc[2][8] = {};
    for (int ks = 0; ks < 8; ++ks) {
      v16bf af[2];
#pragma unroll
      for (int rt = 0; rt < 2; ++rt) {
        const bf16* pp = &P[(r0 + rt * 16 + ml) * 256 + ks * 32];
        af[rt] = frag_cat(pp + 16 * h, pp + 16 * h + 8);
      }
#pragma unroll
      for (int ct = 0; ct < 8; ++ct) {
        const bf16* hp = &Ht[(ct * 16 + ml) * 256 + ks * 32 + 16 * h];
        v16bf bfr = frag_cat(hp, hp + 8);
#pragma unroll
        for (int rt = 0; rt < 2; ++rt) acc[rt][ct] = wmma_bf16(af[rt], bfr, acc[rt][ct]);
      }
    }
#pragma unroll
    for (int rt = 0; rt < 2; ++rt)
#pragma unroll
      for (int ct = 0; ct < 8; ++ct)
#pragma unroll
        for (int e = 0; e < 8; ++e) {
          int grow = r0 + rt * 16 + 8 * h + e;
          int col = cg * 128 + ct * 16 + ml;
          float o = acc[rt][ct][e];
          float gv = gateF[(gRowBase + grow) * HID_ + col];
          attnO[(gRowBase + grow) * 2048 + hh * 512 + col] = (bf16)(silu_f(o) * gv);
        }
    __syncthreads();
  }
}

// ---------------------------------------------------------------- launcher
extern "C" void kernel_launch(void* const* d_in, const int* in_sizes, int n_in,
                              void* d_out, int out_size, void* d_ws, size_t ws_size,
                              hipStream_t stream) {
  const float* x    = (const float*)d_in[0];
  const float* ln_g = (const float*)d_in[1];
  const float* ln_b = (const float*)d_in[2];
  const float* w_qk = (const float*)d_in[3];
  const float* g4   = (const float*)d_in[4];
  const float* b4   = (const float*)d_in[5];
  const float* g2   = (const float*)d_in[6];
  const float* b2   = (const float*)d_in[7];
  const float* w_hi = (const float*)d_in[8];
  const float* w_ga = (const float*)d_in[9];
  const float* w_ou = (const float*)d_in[10];
  float* out = (float*)d_out;
  (void)in_sizes; (void)n_in; (void)out_size; (void)ws_size;

  char* ws = (char*)d_ws;
  size_t off = 0;
  auto alloc = [&](size_t bytes) -> char* {
    char* p = ws + off;
    off += (bytes + 255) & ~(size_t)255;
    return p;
  };
  const long MT = (long)B_ * SEQ_;
  float* xs_f   = (float*)alloc(MT * DIM_ * 4);
  bf16*  xs_b   = (bf16*) alloc(MT * DIM_ * 2);
  float* qk_f   = (float*)alloc(MT * QK_ * 4);
  bf16*  hid_b  = (bf16*) alloc(MT * HID_ * 2);
  bf16*  hid_t  = (bf16*) alloc(MT * HID_ * 2);   // transposed per-group copy
  float* gate_f = (float*)alloc(MT * HID_ * 4);
  float* xg_f   = (float*)alloc(MT * HID_ * 4);
  bf16*  att_b  = (bf16*) alloc(MT * 2048 * 2);
  float* qkm    = (float*)alloc((size_t)B_ * G_ * QK_ * 4);
  bf16*  wqk_t  = (bf16*) alloc(512 * 128 * 2);    // 128 x 512
  bf16*  whi_t  = (bf16*) alloc(512 * 1024 * 2);   // 1024 x 512
  bf16*  wga_t  = (bf16*) alloc(512 * 512 * 2);    // 512 x 512
  bf16*  wou_t  = (bf16*) alloc(2560 * 512 * 2);   // 512 x 2560

  k_cvt_t<<<(512 * 128 + 255) / 256, 256, 0, stream>>>(w_qk, wqk_t, 512, 128);
  k_cvt_t<<<(512 * 1024 + 255) / 256, 256, 0, stream>>>(w_hi, whi_t, 512, 1024);
  k_cvt_t<<<(512 * 512 + 255) / 256, 256, 0, stream>>>(w_ga, wga_t, 512, 512);
  k_cvt_t<<<(2560 * 512 + 255) / 256, 256, 0, stream>>>(w_ou, wou_t, 2560, 512);

  k_ln_shift<<<(unsigned)MT, 256, 0, stream>>>(x, ln_g, ln_b, xs_f, xs_b);

  dim3 gQK(1, 256), gHG(8, 256), gGA(4, 256), gFI(4, 256);
  // qk = xs @ w_qk
  k_gemm_bf16<<<gQK, 256, 0, stream>>>(xs_b, xs_b, DIM_, DIM_, DIM_, wqk_t, 128, 512, 0,
                                       qk_f, nullptr, nullptr, nullptr, nullptr, nullptr, nullptr);
  // [hidden|gate] = xs @ w_hidden (+ transposed hidden copy)
  k_gemm_bf16<<<gHG, 256, 0, stream>>>(xs_b, xs_b, DIM_, DIM_, DIM_, whi_t, 1024, 512, 1,
                                       nullptr, hid_b, hid_t, gate_f, nullptr, nullptr, nullptr);
  // xg = silu(xs @ w_gate)
  k_gemm_bf16<<<gGA, 256, 0, stream>>>(xs_b, xs_b, DIM_, DIM_, DIM_, wga_t, 512, 512, 2,
                                       xg_f, nullptr, nullptr, nullptr, nullptr, nullptr, nullptr);

  k_qkmean<<<B_ * G_, 128, 0, stream>>>(qk_f, qkm);

  // attention: 512 blocks, 192KB dynamic LDS
  k_attn<<<B_ * G_ * 4, 256, 196608, stream>>>(qk_f, qkm, g4, b4, g2, b2, hid_t, gate_f, att_b);

  // out = ([hidden|attn] @ w_out) * xg + xs
  k_gemm_bf16<<<gFI, 256, 0, stream>>>(hid_b, att_b, 512, 512, 2048, wou_t, 512, 2560, 3,
                                       nullptr, nullptr, nullptr, nullptr, xg_f, xs_f, out);
}
